// Decoder_66580583022908
// MI455X (gfx1250) — matrix operand order repaired and checked by reading, not compiled, not run on hardware
//
#include <hip/hip_runtime.h>
#include <hip/hip_bf16.h>

// Problem sizes (fixed by the reference)
#define B_    512
#define T_    256
#define FEAT_ 256
#define HID_  256

typedef __attribute__((ext_vector_type(16))) __bf16 v16bf;
typedef __attribute__((ext_vector_type(8)))  __bf16 v8bf;
typedef __attribute__((ext_vector_type(8)))  float  v8f;

__device__ __forceinline__ float sigmoidf_(float x) { return 1.0f / (1.0f + __expf(-x)); }

// ---------------------------------------------------------------------------
// Small utility kernels
// ---------------------------------------------------------------------------
__global__ void cvt_f32_bf16(const float* __restrict__ src, __bf16* __restrict__ dst, int n) {
    int i = blockIdx.x * 256 + threadIdx.x;
    if (i < n) dst[i] = (__bf16)src[i];
}

__global__ void zero_f32(float* __restrict__ p, int n) {
    int i = blockIdx.x * 256 + threadIdx.x;
    if (i < n) p[i] = 0.0f;
}

__global__ void add_vec(const float* __restrict__ a, const float* __restrict__ b,
                        float* __restrict__ o, int n) {
    int i = blockIdx.x * 256 + threadIdx.x;
    if (i < n) o[i] = a[i] + b[i];
}

// pack [d | c] (fp32) -> dcb (bf16) [B, 2*HID], d in cols 0..255, c in 256..511
__global__ void cvt_dc(const float* __restrict__ d, const float* __restrict__ c,
                       __bf16* __restrict__ dcb) {
    int i = blockIdx.x * 256 + threadIdx.x;   // B*HID threads
    int b = i >> 8, h = i & 255;
    dcb[(size_t)b * (2 * HID_) + h]         = (__bf16)d[i];
    dcb[(size_t)b * (2 * HID_) + HID_ + h]  = (__bf16)c[i];
}

// ---------------------------------------------------------------------------
// WMMA GEMM:  C[M,N](f32 or bf16) = A[M,K](bf16,row-major,lda) @ W[N,K]^T (bf16,ldb) + bias[n]
// One wave -> 16x64 strip of C. K-loop unrolled x2 with register double-buffering:
// each phase issues the NEXT clause of 10 b128 loads before consuming the CURRENT
// fragments with 4 back-to-back v_wmma_f32_16x16x32_bf16, so the pre-WMMA wait
// retires at loadcnt>0 (load latency hidden behind the matrix pipe).
// blockDim = 256 (8 waves); grid = (N/64, M/128). Requires M%128==0, N%64==0, K%64==0.
// ---------------------------------------------------------------------------
__global__ __launch_bounds__(256)
void gemm_bf16_wmma(const __bf16* __restrict__ A, int lda,
                    const __bf16* __restrict__ W, int ldb,
                    float* __restrict__ Cf, __bf16* __restrict__ Cb, int ldc,
                    const float* __restrict__ bias, int K) {
    const int lane  = threadIdx.x & 31;
    const int wv    = threadIdx.x >> 5;
    const int mtile = blockIdx.y * 8 + wv;     // 16-row tile index
    const int n0    = blockIdx.x * 64;         // 4 adjacent 16-col tiles
    const int r     = lane & 15;
    const int kh    = lane >> 4;               // K-half select (0/1)

    // A fragment rows: lane covers K = {kk+kh*8 .. +7} and {kk+16+kh*8 .. +7}
    const __bf16* arow = A + (size_t)(mtile * 16 + r) * lda + kh * 8;
    // B fragment rows: lane = column n, K = {kk+kh*8 .. +7} and {kk+16+kh*8 .. +7}
    const __bf16* brow[4];
#pragma unroll
    for (int nt = 0; nt < 4; ++nt)
        brow[nt] = W + (size_t)(n0 + nt * 16 + r) * ldb + kh * 8;

    v8f acc[4] = {};
    v8bf alo[2], ahi[2], bl[2][4], bh[2][4];

    auto load_set = [&](int s, int kk) {
        alo[s] = *(const v8bf*)(arow + kk);
        ahi[s] = *(const v8bf*)(arow + kk + 16);
#pragma unroll
        for (int nt = 0; nt < 4; ++nt) {
            bl[s][nt] = *(const v8bf*)(brow[nt] + kk);
            bh[s][nt] = *(const v8bf*)(brow[nt] + kk + 16);
        }
    };
    auto wmma_set = [&](int s) {
        v16bf afrag = __builtin_shufflevector(alo[s], ahi[s],
            0, 1, 2, 3, 4, 5, 6, 7, 8, 9, 10, 11, 12, 13, 14, 15);
#pragma unroll
        for (int nt = 0; nt < 4; ++nt) {
            v16bf bfrag = __builtin_shufflevector(bl[s][nt], bh[s][nt],
                0, 1, 2, 3, 4, 5, 6, 7, 8, 9, 10, 11, 12, 13, 14, 15);
            acc[nt] = __builtin_amdgcn_wmma_f32_16x16x32_bf16(
                false, afrag, false, bfrag, (short)0, acc[nt], false, false);
        }
    };

    load_set(0, 0);                               // prologue
    for (int k0 = 0; k0 < K; k0 += 64) {
        load_set(1, k0 + 32);                     // in flight while buffer 0 computes
        __builtin_prefetch(arow + k0 + 64, 0, 1); // global_prefetch_b8 (next strip)
        wmma_set(0);
        if (k0 + 64 < K) load_set(0, k0 + 64);    // in flight while buffer 1 computes
        wmma_set(1);
    }

    // C layout: VGPR v -> row m = mtile*16 + kh*8 + v ; col n = n0 + nt*16 + r
    const int mbase = mtile * 16 + kh * 8;
#pragma unroll
    for (int nt = 0; nt < 4; ++nt) {
        const int n = n0 + nt * 16 + r;
        const float bv = bias ? bias[n] : 0.0f;
#pragma unroll
        for (int v = 0; v < 8; ++v) {
            const size_t idx = (size_t)(mbase + v) * ldc + n;
            const float val = acc[nt][v] + bv;
            if (Cb) Cb[idx] = (__bf16)val;
            else    Cf[idx] = val;
        }
    }
}

// ---------------------------------------------------------------------------
// Fused per-step attention: scores -> softmax -> context -> fc  (one WG per batch b)
//   e[t]   = sum_f tanh(HW[b,t,f] + dcW[b,f]) * W2[f]          (attn_b2 softmax-invariant)
//   beta   = softmax_t(e)
//   ctx[f] = sum_t beta[t] * H[b,t,f]
//   y_tilde[b] = ctx . fcW[0:256] + fcW[256]*Y[b,t] + fc_b
// Lane owns a blocked 8-feature slice -> every global access is a b128 load.
// ---------------------------------------------------------------------------
__global__ __launch_bounds__(256)
void attn_step(const __bf16* __restrict__ HWb, const __bf16* __restrict__ Hb,
               const float* __restrict__ dcW, const float* __restrict__ W2,
               const float* __restrict__ fcW, const float* __restrict__ fcb,
               const float* __restrict__ Y, int t_step,
               float* __restrict__ ctx, float* __restrict__ y_tilde) {
    __shared__ float e_s[T_];
    __shared__ float red_s[8];
    __shared__ float ctx_s[8][FEAT_];

    const int b = blockIdx.x;
    const int tid = threadIdx.x;
    const int lane = tid & 31;
    const int wv = tid >> 5;
    const int f0 = lane * 8;                 // blocked feature slice per lane

    // Register-resident per-lane weight slices (reused across all 256 timesteps).
    float dcw_r[8], w2_r[8];
#pragma unroll
    for (int j = 0; j < 8; ++j) {
        dcw_r[j] = dcW[(size_t)b * FEAT_ + f0 + j];
        w2_r[j]  = W2[f0 + j];
    }

    // ---- scores: wave wv handles timesteps t = wv, wv+8, ...; one b128 load per row
    for (int t = wv; t < T_; t += 8) {
        const v8bf hw = *(const v8bf*)(HWb + ((size_t)b * T_ + t) * FEAT_ + f0);
        float acc = 0.0f;
#pragma unroll
        for (int j = 0; j < 8; ++j)
            acc += tanhf((float)hw[j] + dcw_r[j]) * w2_r[j];
#pragma unroll
        for (int off = 16; off > 0; off >>= 1) acc += __shfl_xor(acc, off, 32);
        if (lane == 0) e_s[t] = acc;
    }
    __syncthreads();

    // ---- softmax over 256 timesteps (beta stored normalized, in place)
    const float v = e_s[tid];
    float m = v;
#pragma unroll
    for (int off = 16; off > 0; off >>= 1) m = fmaxf(m, __shfl_xor(m, off, 32));
    if (lane == 0) red_s[wv] = m;
    __syncthreads();
    if (tid == 0) {
        float mm = red_s[0];
        for (int i = 1; i < 8; ++i) mm = fmaxf(mm, red_s[i]);
        red_s[0] = mm;
    }
    __syncthreads();
    const float gmax = red_s[0];
    const float ex = __expf(v - gmax);
    float s = ex;
#pragma unroll
    for (int off = 16; off > 0; off >>= 1) s += __shfl_xor(s, off, 32);
    __syncthreads();                      // all waves have consumed red_s[0] (gmax)
    if (lane == 0) red_s[wv] = s;
    __syncthreads();
    if (tid == 0) {
        float ss = 0.0f;
        for (int i = 0; i < 8; ++i) ss += red_s[i];
        red_s[0] = ss;
    }
    __syncthreads();
    const float inv = 1.0f / red_s[0];
    e_s[tid] = ex * inv;                  // normalized beta
    __syncthreads();

    // ---- context: wave-parallel over t, b128 loads, 8-way LDS partial reduce
    float cacc[8] = {};
    for (int t = wv; t < T_; t += 8) {
        const float beta = e_s[t];
        const v8bf hrow = *(const v8bf*)(Hb + ((size_t)b * T_ + t) * FEAT_ + f0);
#pragma unroll
        for (int j = 0; j < 8; ++j) cacc[j] += beta * (float)hrow[j];
    }
#pragma unroll
    for (int j = 0; j < 8; ++j) ctx_s[wv][f0 + j] = cacc[j];
    __syncthreads();

    float cv = 0.0f;
#pragma unroll
    for (int w = 0; w < 8; ++w) cv += ctx_s[w][tid];
    ctx[(size_t)b * FEAT_ + tid] = cv;

    // ---- fc: y_tilde[b]
    float pr = cv * fcW[tid];
#pragma unroll
    for (int off = 16; off > 0; off >>= 1) pr += __shfl_xor(pr, off, 32);
    __syncthreads();                      // red_s[0] (inv) consumed long ago; safe to reuse
    if (lane == 0) red_s[wv] = pr;
    __syncthreads();
    if (tid == 0) {
        float tot = 0.0f;
        for (int i = 0; i < 8; ++i) tot += red_s[i];
        y_tilde[b] = tot + fcW[FEAT_] * Y[(size_t)b * T_ + t_step] + fcb[0];
    }
}

// ---------------------------------------------------------------------------
// LSTM cell elementwise update. gatesW = d @ Whh^T + (bih+bhh) precomputed.
// gates += y_tilde * Wih ; PyTorch gate order i,f,g,o.
// ---------------------------------------------------------------------------
__global__ void lstm_cell(const float* __restrict__ gatesW, const float* __restrict__ y_tilde,
                          const float* __restrict__ Wih,
                          float* __restrict__ d, float* __restrict__ c) {
    int i = blockIdx.x * 256 + threadIdx.x;   // B*HID
    int b = i >> 8, h = i & 255;
    const float yt = y_tilde[b];
    const float* g = gatesW + (size_t)b * 4 * HID_;
    const float gi = g[h]            + yt * Wih[h];
    const float gf = g[HID_ + h]     + yt * Wih[HID_ + h];
    const float gg = g[2 * HID_ + h] + yt * Wih[2 * HID_ + h];
    const float go = g[3 * HID_ + h] + yt * Wih[3 * HID_ + h];
    const float cn = sigmoidf_(gf) * c[i] + sigmoidf_(gi) * tanhf(gg);
    const float dn = sigmoidf_(go) * tanhf(cn);
    c[i] = cn;
    d[i] = dn;
}

// ---------------------------------------------------------------------------
// y_pred[b] = [d, ctx] . fcfW + fcf_b
// ---------------------------------------------------------------------------
__global__ __launch_bounds__(256)
void final_fc(const float* __restrict__ d, const float* __restrict__ ctx,
              const float* __restrict__ fcfW, const float* __restrict__ fcfb,
              float* __restrict__ out) {
    __shared__ float red_s[8];
    const int b = blockIdx.x, tid = threadIdx.x, lane = tid & 31, wv = tid >> 5;
    float p = d[(size_t)b * HID_ + tid] * fcfW[tid]
            + ctx[(size_t)b * FEAT_ + tid] * fcfW[HID_ + tid];
#pragma unroll
    for (int off = 16; off > 0; off >>= 1) p += __shfl_xor(p, off, 32);
    if (lane == 0) red_s[wv] = p;
    __syncthreads();
    if (tid == 0) {
        float tot = 0.0f;
        for (int i = 0; i < 8; ++i) tot += red_s[i];
        out[b] = tot + fcfb[0];
    }
}

// ---------------------------------------------------------------------------
extern "C" void kernel_launch(void* const* d_in, const int* in_sizes, int n_in,
                              void* d_out, int out_size, void* d_ws, size_t ws_size,
                              hipStream_t stream) {
    (void)in_sizes; (void)n_in; (void)out_size; (void)ws_size;

    const float* H    = (const float*)d_in[0];
    const float* Y    = (const float*)d_in[1];
    const float* aW1  = (const float*)d_in[2];   // [256, 768] : [d | c | H] columns
    const float* ab1  = (const float*)d_in[3];
    const float* aW2  = (const float*)d_in[4];
    /* ab2 (d_in[5]) is softmax-invariant -> unused */
    const float* Wih  = (const float*)d_in[6];   // [1024, 1]
    const float* Whh  = (const float*)d_in[7];   // [1024, 256]
    const float* bih  = (const float*)d_in[8];
    const float* bhh  = (const float*)d_in[9];
    const float* fcW  = (const float*)d_in[10];  // [1, 257]
    const float* fcb  = (const float*)d_in[11];
    const float* fcfW = (const float*)d_in[12];  // [1, 512]
    const float* fcfb = (const float*)d_in[13];
    float* out = (float*)d_out;

    // bump allocator over workspace (~140 MB total)
    char* p = (char*)d_ws;
    auto alloc = [&](size_t bytes) -> char* {
        char* r = p;
        p += (bytes + 255) & ~(size_t)255;
        return r;
    };
    __bf16* Hb     = (__bf16*)alloc((size_t)B_ * T_ * FEAT_ * 2);       // 64 MB
    __bf16* HWb    = (__bf16*)alloc((size_t)B_ * T_ * FEAT_ * 2);       // 64 MB (L2-resident)
    __bf16* W1b    = (__bf16*)alloc((size_t)FEAT_ * (2 * HID_ + FEAT_) * 2);
    __bf16* Whhb   = (__bf16*)alloc((size_t)4 * HID_ * HID_ * 2);
    float*  gbias  = (float*) alloc((size_t)4 * HID_ * 4);
    float*  dS     = (float*) alloc((size_t)B_ * HID_ * 4);
    float*  cS     = (float*) alloc((size_t)B_ * HID_ * 4);
    __bf16* dcb    = (__bf16*)alloc((size_t)B_ * 2 * HID_ * 2);
    float*  dcW    = (float*) alloc((size_t)B_ * FEAT_ * 4);
    float*  gatesW = (float*) alloc((size_t)B_ * 4 * HID_ * 4);
    float*  ctx    = (float*) alloc((size_t)B_ * FEAT_ * 4);
    float*  ytl    = (float*) alloc((size_t)B_ * 4);

    // ---- one-time setup
    const int nH = B_ * T_ * FEAT_;
    cvt_f32_bf16<<<(nH + 255) / 256, 256, 0, stream>>>(H, Hb, nH);
    const int nW1 = FEAT_ * (2 * HID_ + FEAT_);
    cvt_f32_bf16<<<(nW1 + 255) / 256, 256, 0, stream>>>(aW1, W1b, nW1);
    const int nWhh = 4 * HID_ * HID_;
    cvt_f32_bf16<<<(nWhh + 255) / 256, 256, 0, stream>>>(Whh, Whhb, nWhh);
    add_vec<<<(4 * HID_ + 255) / 256, 256, 0, stream>>>(bih, bhh, gbias, 4 * HID_);
    zero_f32<<<(B_ * HID_ + 255) / 256, 256, 0, stream>>>(dS, B_ * HID_);
    zero_f32<<<(B_ * HID_ + 255) / 256, 256, 0, stream>>>(cS, B_ * HID_);

    // Hoisted time-invariant GEMM: HW[b,t,:] = H[b,t,:] @ W1[:,512:768]^T  (bf16 out)
    {
        dim3 grid(FEAT_ / 64, (B_ * T_) / 128);
        gemm_bf16_wmma<<<grid, 256, 0, stream>>>(Hb, FEAT_, W1b + 2 * HID_, 2 * HID_ + FEAT_,
                                                 nullptr, HWb, FEAT_, nullptr, FEAT_);
    }

    // ---- sequential scan over T timesteps
    for (int t = 0; t < T_; ++t) {
        cvt_dc<<<(B_ * HID_ + 255) / 256, 256, 0, stream>>>(dS, cS, dcb);
        // gates = d @ Whh^T + (bih+bhh)   [512 x 1024], K=256 (d = first half of dcb)
        {
            dim3 grid((4 * HID_) / 64, B_ / 128);
            gemm_bf16_wmma<<<grid, 256, 0, stream>>>(dcb, 2 * HID_, Whhb, HID_,
                                                     gatesW, nullptr, 4 * HID_, gbias, HID_);
        }
        // dcW = [d|c] @ W1[:,0:512]^T + b1   [512 x 256], K=512
        {
            dim3 grid(FEAT_ / 64, B_ / 128);
            gemm_bf16_wmma<<<grid, 256, 0, stream>>>(dcb, 2 * HID_, W1b, 2 * HID_ + FEAT_,
                                                     dcW, nullptr, FEAT_, ab1, 2 * HID_);
        }
        attn_step<<<B_, 256, 0, stream>>>(HWb, Hb, dcW, aW2, fcW, fcb, Y, t, ctx, ytl);
        lstm_cell<<<(B_ * HID_ + 255) / 256, 256, 0, stream>>>(gatesW, ytl, Wih, dS, cS);
    }

    final_fc<<<B_, 256, 0, stream>>>(dS, ctx, fcfW, fcfb, out);
}
